// TriangleAttn_26293789786412
// MI455X (gfx1250) — matrix operand order repaired and checked by reading, not compiled, not run on hardware
//
#include <hip/hip_runtime.h>
#include <hip/hip_bf16.h>
#include <math.h>

#define N    256
#define D    128
#define HH   4
#define DHH  32
#define NROW (N * N) // 65536

typedef __attribute__((ext_vector_type(16))) _Float16 v16h;
typedef __attribute__((ext_vector_type(8)))  _Float16 v8h;
typedef __attribute__((ext_vector_type(8)))  float    v8f;
typedef __attribute__((ext_vector_type(2)))  float    v2f;
typedef __attribute__((ext_vector_type(4)))  unsigned int uint4v;
typedef __attribute__((ext_vector_type(8)))  int      int8v;
typedef __attribute__((ext_vector_type(4)))  int      int4v;

#define NEGF (-3.402823466e38f)

// A fragment, 16-bit, 16x32 (MxK): lane holds row m=L%16.
// elems j = row[half*16 + j] (contiguous)
__device__ __forceinline__ v16h load_afrag(const _Float16* rowp, int half) {
  v8h lo = *(const v8h*)(rowp + half * 16);
  v8h hi = *(const v8h*)(rowp + half * 16 + 8);
  return __builtin_shufflevector(lo, hi, 0,1,2,3,4,5,6,7,8,9,10,11,12,13,14,15);
}

// B fragment, 16-bit, 32x16 (KxN): lane holds col n=L%16.
// From K-major storage (colp = &Bt[n][0]): elems j = colp[half*16 + j] (contiguous)
__device__ __forceinline__ v16h load_bfrag(const _Float16* colp, int half) {
  v8h lo = *(const v8h*)(colp + half * 16);
  v8h hi = *(const v8h*)(colp + half * 16 + 8);
  return __builtin_shufflevector(lo, hi, 0,1,2,3,4,5,6,7,8,9,10,11,12,13,14,15);
}

__device__ __forceinline__ v8f wmma_f16(v16h a, v16h b, v8f c) {
  return __builtin_amdgcn_wmma_f32_16x16x32_f16(false, a, false, b, (short)0, c,
                                                false, false);
}

// Issue one TDM 2D tile load (uniform args): tile_d0 x tile_d1 elements of 2B,
// row stride `stride0` elements, into LDS at byte offset lds_addr.
// clang-23 toolchain: 6-arg builtin (g0, g1, g2, g3, g4, cpol).
__device__ __forceinline__ void tdm_load_2d_f16(unsigned long long gaddr,
                                                unsigned lds_addr,
                                                unsigned tile_d0, unsigned tile_d1,
                                                unsigned stride0) {
  uint4v g0;
  g0[0] = 1u;                                         // count=1, user mode
  g0[1] = lds_addr;                                   // lds_addr [63:32]
  g0[2] = (unsigned)(gaddr & 0xFFFFFFFFu);            // global_addr[31:0]
  g0[3] = (unsigned)((gaddr >> 32) & 0x01FFFFFFu)     // global_addr[56:32]
          | (2u << 30);                               // type=2 ("image")
  int8v g1;
  g1[0] = (int)(1u << 16);                            // data_size=1 (2 bytes)
  g1[1] = (int)(tile_d0 << 16);                       // tensor_dim0[15:0] @ bit48
  g1[2] = (int)(tile_d1 << 16);                       // tensor_dim1[15:0] @ bit80
  g1[3] = (int)(tile_d0 << 16);                       // tile_dim0 @ bit112
  g1[4] = (int)tile_d1;                               // tile_dim1 @ bit128
  g1[5] = (int)stride0;                               // tensor_dim0_stride[31:0]
  g1[6] = 0;
  g1[7] = 0;
  int4v z4 = {0, 0, 0, 0};                            // groups 2/3 unused (2D)
  int8v z8 = {0, 0, 0, 0, 0, 0, 0, 0};
  __builtin_amdgcn_tensor_load_to_lds(g0, g1, z4, z4, z8, 0);
}

// ---------------------------------------------------------------------------
// Kernel 0: transpose weights into K-major layouts for contiguous B fragments
// ---------------------------------------------------------------------------
__global__ void k_prep(const float* __restrict__ Wqkv, const float* __restrict__ Wg,
                       const float* __restrict__ Wout, _Float16* __restrict__ WtA,
                       float* __restrict__ Wot) {
  int idx = blockIdx.x * 256 + threadIdx.x;
  if (idx < 512 * 128) {
    int c = idx >> 7, k = idx & 127;
    float w = (c < 384) ? Wqkv[k * 384 + c] : Wg[k * 128 + (c - 384)];
    WtA[c * 128 + k] = (_Float16)w;
  }
  int idx2 = idx - 512 * 128;
  if (idx2 >= 0 && idx2 < 128 * 128) {
    int n = idx2 >> 7, k = idx2 & 127;
    Wot[n * 128 + k] = Wout[k * 128 + n];
  }
}

// ---------------------------------------------------------------------------
// Kernel 1: LayerNorm + fused projections (QKV, gate, bias) via WMMA f16
// ---------------------------------------------------------------------------
__global__ void k_lnproj(const float* __restrict__ edges,
                         const float* __restrict__ gamma,
                         const float* __restrict__ beta,
                         const float* __restrict__ Wb,
                         const float* __restrict__ bg,
                         const _Float16* __restrict__ WtA,
                         _Float16* __restrict__ qkv,
                         _Float16* __restrict__ gbuf,
                         float* __restrict__ biasbuf) {
  __shared__ alignas(16) _Float16 xs[16][136];
  const int tid = threadIdx.x;
  const int r0 = blockIdx.x * 16;

  if (tid < 16) {
    const float* xp = edges + (size_t)(r0 + tid) * D;
    float s = 0.f, s2 = 0.f;
    for (int c = 0; c < D; ++c) { float v = xp[c]; s += v; s2 += v * v; }
    float mu = s * (1.0f / D);
    float var = s2 * (1.0f / D) - mu * mu;
    float rs = rsqrtf(var + 1e-5f);
    for (int c = 0; c < D; ++c)
      xs[tid][c] = (_Float16)((xp[c] - mu) * rs * gamma[c] + beta[c]);
  }
  __syncthreads();

  const int wave = tid >> 5, lane = tid & 31;
  const int l16 = lane & 15, half = lane >> 4;

  v16h a[4];
#pragma unroll
  for (int kc = 0; kc < 4; ++kc) a[kc] = load_afrag(&xs[l16][kc * 32], half);

  for (int t = wave * 8; t < wave * 8 + 8; ++t) {
    const _Float16* wrow = WtA + (size_t)(t * 16 + l16) * D;
    v16h bfr[4];
#pragma unroll
    for (int kc = 0; kc < 4; ++kc) bfr[kc] = load_bfrag(wrow + kc * 32, half);
    v8f acc = {0.f, 0.f, 0.f, 0.f, 0.f, 0.f, 0.f, 0.f};
#pragma unroll
    for (int kc = 0; kc < 4; ++kc) acc = wmma_f16(a[kc], bfr[kc], acc);

    if (t < 24) { // qkv columns
#pragma unroll
      for (int r = 0; r < 8; ++r) {
        int row = r0 + r + 8 * half;
        qkv[(size_t)row * 384 + t * 16 + l16] = (_Float16)acc[r];
      }
    } else {      // gate columns: sigmoid(x@Wg + bg) via v_rcp_f32
      int c2 = (t - 24) * 16 + l16;
      float bb = bg[c2];
#pragma unroll
      for (int r = 0; r < 8; ++r) {
        int row = r0 + r + 8 * half;
        float v = acc[r] + bb;
        float sg = __builtin_amdgcn_rcpf(1.0f + __expf(-v));
        gbuf[(size_t)row * D + c2] = (_Float16)sg;
      }
    }
  }

  if (tid < 64) { // bias head: 16 rows x 4 heads, narrow -> scalar dot
    int rr = tid >> 2, hh = tid & 3;
    float dot = 0.f;
    for (int c = 0; c < D; ++c) dot += (float)xs[rr][c] * Wb[c * HH + hh];
    biasbuf[(size_t)(r0 + rr) * HH + hh] = dot;
  }
}

// ---------------------------------------------------------------------------
// Kernel 2: attention for one (h, i, 64-row j block); 4 waves, 16 j-rows/wave
//   K/V staged by the Tensor Data Mover; LDS: Vs 16KB | Ks 16KB (reused as P)
// ---------------------------------------------------------------------------
__global__ void k_attn(const _Float16* __restrict__ qkv,
                       const _Float16* __restrict__ gbuf,
                       const float* __restrict__ biasbuf,
                       const unsigned char* __restrict__ mask,
                       float* __restrict__ attnbuf) {
  __shared__ alignas(16) char smem[49152];
  _Float16* Vs = (_Float16*)smem;              // [256][32], LDS offset 0
  _Float16* Ks = (_Float16*)(smem + 16384);    // [256][32], LDS offset 16384
  _Float16* Ps = (_Float16*)(smem + 16384);    // [4][16][256], reuses Ks region

  const int tid = threadIdx.x;
  const int i = blockIdx.y, h = blockIdx.z;

  // TDM: DMA K_i and V_i tiles (256 rows x 32 halves, row stride 384 halves).
  // TDM ignores EXEC -> must be reached by exactly one wave via branch.
  if (tid == 0) {
    unsigned long long base =
        (unsigned long long)(const void*)qkv +
        ((unsigned long long)(i * N) * 384ull + (unsigned long long)h * DHH) * 2ull;
    tdm_load_2d_f16(base + 128ull * 2ull, 16384u, DHH, N, 384u); // K -> Ks
    tdm_load_2d_f16(base + 256ull * 2ull, 0u,     DHH, N, 384u); // V -> Vs
    __builtin_amdgcn_s_wait_tensorcnt(0);
  }
  __syncthreads();

  const int wave = tid >> 5, lane = tid & 31;
  const int l16 = lane & 15, half = lane >> 4;
  const int j0 = blockIdx.x * 64 + wave * 16;

  const _Float16* qrow = qkv + (size_t)(i * N + j0 + l16) * 384 + h * DHH;
  __builtin_prefetch(qrow, 0, 0); // -> global_prefetch_b8
  v16h aq = load_afrag(qrow, half);

  bool mrow[8];
#pragma unroll
  for (int r = 0; r < 8; ++r)
    mrow[r] = mask[(size_t)i * N + j0 + r + 8 * half] != 0;

  // S = Q K^T : 16 column tiles, DH=32 -> one WMMA each
  v8f S[16];
#pragma unroll
  for (int kt = 0; kt < 16; ++kt) {
    v8f z = {0.f, 0.f, 0.f, 0.f, 0.f, 0.f, 0.f, 0.f};
    S[kt] = wmma_f16(aq, load_bfrag(Ks + (kt * 16 + l16) * DHH, half), z);
  }

  const float scale = 0.17677669529663687f; // 32^-0.5
  float rmax[8], rsum[8], rinv[8];
#pragma unroll
  for (int r = 0; r < 8; ++r) rmax[r] = NEGF;
#pragma unroll
  for (int kt = 0; kt < 16; ++kt) {
#pragma unroll
    for (int r = 0; r < 8; ++r) {
      int jrow = j0 + r + 8 * half;
      float sv = S[kt][r] * scale +
                 biasbuf[((size_t)jrow * N + kt * 16 + l16) * HH + h];
      if (!mrow[r]) sv = NEGF; // masked row: uniform softmax, as in reference
      S[kt][r] = sv;
      rmax[r] = fmaxf(rmax[r], sv);
    }
  }
#pragma unroll
  for (int r = 0; r < 8; ++r) {
    float v = rmax[r];
#pragma unroll
    for (int off = 8; off >= 1; off >>= 1) v = fmaxf(v, __shfl_xor(v, off, 32));
    rmax[r] = v;
  }
#pragma unroll
  for (int r = 0; r < 8; ++r) rsum[r] = 0.f;
#pragma unroll
  for (int kt = 0; kt < 16; ++kt) {
#pragma unroll
    for (int r = 0; r < 8; ++r) {
      float e = __expf(S[kt][r] - rmax[r]);
      S[kt][r] = e;
      rsum[r] += e;
    }
  }
#pragma unroll
  for (int r = 0; r < 8; ++r) {
    float v = rsum[r];
#pragma unroll
    for (int off = 8; off >= 1; off >>= 1) v += __shfl_xor(v, off, 32);
    rinv[r] = __builtin_amdgcn_rcpf(v);
  }

  __syncthreads(); // all waves finished reading Ks; its LDS becomes P storage

  _Float16* myP = Ps + wave * 16 * 256; // [16][256] row-major (j-local, k)
#pragma unroll
  for (int kt = 0; kt < 16; ++kt)
#pragma unroll
    for (int r = 0; r < 8; ++r)
      myP[(r + 8 * half) * 256 + kt * 16 + l16] = (_Float16)(S[kt][r] * rinv[r]);

  __syncthreads();

  // O = P @ V : K=256 in 8 chunks of 32, N=32 as two 16-col tiles
  v8f O0 = {0.f, 0.f, 0.f, 0.f, 0.f, 0.f, 0.f, 0.f};
  v8f O1 = {0.f, 0.f, 0.f, 0.f, 0.f, 0.f, 0.f, 0.f};
#pragma unroll
  for (int kc = 0; kc < 8; ++kc) {
    v16h ap = load_afrag(myP + l16 * 256 + kc * 32, half);
    v16h b0, b1;
#pragma unroll
    for (int j = 0; j < 16; ++j) {
      int kk = kc * 32 + half * 16 + j;
      b0[j] = Vs[kk * DHH + l16];
      b1[j] = Vs[kk * DHH + 16 + l16];
    }
    O0 = wmma_f16(ap, b0, O0);
    O1 = wmma_f16(ap, b1, O1);
  }

  // gate (masked -> finfo.min) and store f32 for the f32 out-projection
#pragma unroll
  for (int r = 0; r < 8; ++r) {
    int grow = i * N + j0 + r + 8 * half;
#pragma unroll
    for (int nt = 0; nt < 2; ++nt) {
      int col = h * DHH + nt * 16 + l16;
      float ov = nt ? O1[r] : O0[r];
      float gv = mrow[r] ? (float)gbuf[(size_t)grow * D + col] : NEGF;
      attnbuf[(size_t)grow * D + col] = gv * ov;
    }
  }
}

// ---------------------------------------------------------------------------
// Kernel 3: out = attn @ W_out + b_out, full-f32 WMMA (16x16x4, 32 K-steps)
// ---------------------------------------------------------------------------
__global__ void k_outproj(const float* __restrict__ attnbuf,
                          const float* __restrict__ Wot,
                          const float* __restrict__ bout,
                          float* __restrict__ out) {
  const int tid = threadIdx.x;
  const int wave = tid >> 5, lane = tid & 31;
  const int l16 = lane & 15, half = lane >> 4;
  const int tile = blockIdx.x * 4 + wave;
  const int r0 = (tile >> 3) * 16, c0 = (tile & 7) * 16;

  const float* arow = attnbuf + (size_t)(r0 + l16) * D; // A row m = l16
  const float* brow = Wot + (size_t)(c0 + l16) * D;     // B col n = l16 (K-major)

  v8f acc = {0.f, 0.f, 0.f, 0.f, 0.f, 0.f, 0.f, 0.f};
#pragma unroll
  for (int kb = 0; kb < 32; ++kb) {
    int kk = kb * 4 + 2 * half; // f32 frag: VGPR0 K=2*half, VGPR1 K=2*half+1
    v2f av = {arow[kk], arow[kk + 1]};
    v2f bv = {brow[kk], brow[kk + 1]};
    acc = __builtin_amdgcn_wmma_f32_16x16x4_f32(false, av, false, bv, (short)0,
                                                acc, false, false);
  }
  float bb = bout[c0 + l16];
#pragma unroll
  for (int r = 0; r < 8; ++r)
    out[(size_t)(r0 + r + 8 * half) * D + c0 + l16] = acc[r] + bb;
}

// ---------------------------------------------------------------------------
extern "C" void kernel_launch(void* const* d_in, const int* in_sizes, int n_in,
                              void* d_out, int out_size, void* d_ws, size_t ws_size,
                              hipStream_t stream) {
  (void)in_sizes; (void)n_in; (void)out_size; (void)ws_size;
  const float* edges        = (const float*)d_in[0];
  const unsigned char* mask = (const unsigned char*)d_in[1]; // jnp bool = 1 byte
  const float* gamma        = (const float*)d_in[2];
  const float* beta         = (const float*)d_in[3];
  const float* Wqkv         = (const float*)d_in[4];
  const float* Wb           = (const float*)d_in[5];
  const float* Wg           = (const float*)d_in[6];
  const float* bg           = (const float*)d_in[7];
  const float* Wout         = (const float*)d_in[8];
  const float* bout         = (const float*)d_in[9];

  char* ws = (char*)d_ws;
  _Float16* qkv  = (_Float16*)ws; ws += (size_t)NROW * 384 * 2; // 48 MB
  _Float16* gbuf = (_Float16*)ws; ws += (size_t)NROW * D * 2;   // 16 MB
  float* biasbuf = (float*)ws;    ws += (size_t)NROW * HH * 4;  //  1 MB
  float* attnbuf = (float*)ws;    ws += (size_t)NROW * D * 4;   // 32 MB
  _Float16* WtA  = (_Float16*)ws; ws += (size_t)512 * D * 2;
  float* Wot     = (float*)ws;    ws += (size_t)D * D * 4;

  k_prep<<<dim3(320), dim3(256), 0, stream>>>(Wqkv, Wg, Wout, WtA, Wot);
  k_lnproj<<<dim3(NROW / 16), dim3(128), 0, stream>>>(edges, gamma, beta, Wb, bg,
                                                      WtA, qkv, gbuf, biasbuf);
  k_attn<<<dim3(4, N, HH), dim3(128), 0, stream>>>(qkv, gbuf, biasbuf, mask,
                                                   attnbuf);
  k_outproj<<<dim3((NROW / 16) * 8 / 4), dim3(128), 0, stream>>>(attnbuf, Wot,
                                                                 bout,
                                                                 (float*)d_out);
}
